// CMGAutoencoder_90117003805173
// MI455X (gfx1250) — matrix operand order repaired and checked by reading, not compiled, not run on hardware
//
#include <hip/hip_runtime.h>
#include <hip/hip_bf16.h>

#define NN 50000
#define FF 1433
#define HH 64
#define EE 1600000
#define CC 5000

typedef __attribute__((ext_vector_type(16))) __bf16 v16bf;
typedef __attribute__((ext_vector_type(8)))  float  v8f;
typedef __attribute__((ext_vector_type(8)))  unsigned int v8u;

// ---- helpers -------------------------------------------------------------

// Pack two floats to bf16x2 via native conversion (lowers to v_cvt_pk_bf16_f32).
static __device__ __forceinline__ unsigned pk2(float lo, float hi) {
    __bf16 a = (__bf16)lo;
    __bf16 b = (__bf16)hi;
    unsigned short ua = __builtin_bit_cast(unsigned short, a);
    unsigned short ub = __builtin_bit_cast(unsigned short, b);
    return (unsigned)ua | ((unsigned)ub << 16);
}

static __device__ __forceinline__ v16bf load_frag8(const unsigned* p) {
    v8u b;
#pragma unroll
    for (int i = 0; i < 8; ++i) b[i] = p[i];
    return __builtin_bit_cast(v16bf, b);
}

// ---- encoder GEMM: h0 = x @ W_enc  (bf16 WMMA, f32 accum) ---------------
// block = 256 threads = 8 waves; wave -> 16 rows x 64 cols; block -> 128 rows.
// Both x-tile (128x32) and W-tile (32x64) staged in LDS as packed bf16.

__global__ __launch_bounds__(256) void enc_gemm_kernel(
        const float* __restrict__ x, const float* __restrict__ W,
        float* __restrict__ h0) {
    __shared__ unsigned sX[128 * 16];  // 128 rows x 32 K (bf16 pairs) = 8 KB
    __shared__ unsigned sW[32 * 32];   // 64 N x 32 K (bf16 K-pairs)   = 4 KB
    const int tid   = threadIdx.x;
    const int wave  = tid >> 5;
    const int lane  = tid & 31;
    const int khalf = lane >> 4;
    const int lrow  = lane & 15;
    const int row0  = blockIdx.x * 128;

    v8f acc0 = {}, acc1 = {}, acc2 = {}, acc3 = {};

    for (int k0 = 0; k0 < FF; k0 += 32) {
        __syncthreads();
        if (k0 + 32 <= FF) {
            // fast path: no K guards, branch-free
#pragma unroll
            for (int j = 0; j < 4; ++j) {           // W tile: 1024 uints
                int u   = tid + j * 256;
                int nn  = u >> 4, kc = u & 15;
                const float* p = W + (k0 + 2 * kc) * HH + nn;
                sW[u] = pk2(p[0], p[HH]);
            }
#pragma unroll
            for (int j = 0; j < 8; ++j) {           // x tile: 2048 uints
                int u = tid + j * 256;
                int r = u >> 4, c = u & 15;
                int row = row0 + r;
                if (row > NN - 1) row = NN - 1;     // clamp; stores are guarded
                const float* p = x + (long)row * FF + k0 + c * 2;
                sX[u] = pk2(p[0], p[1]);
                if (j == 0 && k0 + 64 <= FF) __builtin_prefetch(p + 32, 0, 3);
            }
        } else {
            // K tail (k0 = 1408, 25 valid columns)
#pragma unroll
            for (int j = 0; j < 4; ++j) {
                int u   = tid + j * 256;
                int nn  = u >> 4, kc = u & 15;
                int k   = k0 + 2 * kc;
                float f0 = (k < FF) ? W[k * HH + nn] : 0.f;
                float f1 = (k + 1 < FF) ? W[(k + 1) * HH + nn] : 0.f;
                sW[u] = pk2(f0, f1);
            }
#pragma unroll
            for (int j = 0; j < 8; ++j) {
                int u = tid + j * 256;
                int r = u >> 4, c = u & 15;
                int row = row0 + r;
                if (row > NN - 1) row = NN - 1;
                const float* p = x + (long)row * FF;
                int ka = k0 + c * 2, kb = ka + 1;
                float f0 = (ka < FF) ? p[ka] : 0.f;
                float f1 = (kb < FF) ? p[kb] : 0.f;
                sX[u] = pk2(f0, f1);
            }
        }
        __syncthreads();

        // A fragment: two ds_load_b128 per lane
        const int abase = (wave * 16 + lrow) * 16 + khalf * 8;
        v8u av;
#pragma unroll
        for (int i = 0; i < 4; ++i) av[i]     = sX[abase + i];
#pragma unroll
        for (int i = 0; i < 4; ++i) av[4 + i] = sX[abase + 4 + i];
        v16bf a = __builtin_bit_cast(v16bf, av);

        const unsigned* wrow = &sW[lrow * 16 + khalf * 8];
        v16bf b0 = load_frag8(wrow +   0);
        v16bf b1 = load_frag8(wrow + 256);
        v16bf b2 = load_frag8(wrow + 512);
        v16bf b3 = load_frag8(wrow + 768);
        acc0 = __builtin_amdgcn_wmma_f32_16x16x32_bf16(false, a, false, b0, (short)0, acc0, false, false);
        acc1 = __builtin_amdgcn_wmma_f32_16x16x32_bf16(false, a, false, b1, (short)0, acc1, false, false);
        acc2 = __builtin_amdgcn_wmma_f32_16x16x32_bf16(false, a, false, b2, (short)0, acc2, false, false);
        acc3 = __builtin_amdgcn_wmma_f32_16x16x32_bf16(false, a, false, b3, (short)0, acc3, false, false);
    }

    // C layout: VGPR r -> row 8*khalf + r, col lane%16
    const int srow = row0 + wave * 16 + 8 * khalf;
#pragma unroll
    for (int r = 0; r < 8; ++r) {
        int rr = srow + r;
        if (rr < NN) {
            float* orow = h0 + (long)rr * HH;
            orow[lrow +  0] = acc0[r];
            orow[lrow + 16] = acc1[r];
            orow[lrow + 32] = acc2[r];
            orow[lrow + 48] = acc3[r];
        }
    }
}

// ---- decoder GEMM: x_d = agg_c @ W_dec + b_dec  (bf16 WMMA) -------------

__global__ __launch_bounds__(256) void dec_gemm_kernel(
        const float* __restrict__ A, const float* __restrict__ W,
        const float* __restrict__ bias, float* __restrict__ xd) {
    __shared__ unsigned sW[64 * 32];   // 64 cols of this N-slice x K=64 (bf16 K-pairs) = 8 KB
    const int tid   = threadIdx.x;
    const int wave  = tid >> 5;
    const int lane  = tid & 31;
    const int khalf = lane >> 4;
    const int lrow  = lane & 15;
    const int row0  = blockIdx.x * 128 + wave * 16;
    const int col0  = blockIdx.y * 64;
    int m = row0 + lrow;
    if (m > CC - 1) m = CC - 1;        // clamp; stores are guarded

    if (col0 + 64 <= FF) {
#pragma unroll
        for (int j = 0; j < 8; ++j) {  // 2048 uints
            int u   = tid + j * 256;
            int nn  = u >> 5, kc = u & 31;
            const float* p = W + (2 * kc) * FF + col0 + nn;
            sW[u] = pk2(p[0], p[FF]);
        }
    } else {
#pragma unroll
        for (int j = 0; j < 8; ++j) {
            int u   = tid + j * 256;
            int nn  = u >> 5, kc = u & 31;
            int n   = col0 + nn;
            float f0 = (n < FF) ? W[(2 * kc) * FF + n] : 0.f;
            float f1 = (n < FF) ? W[(2 * kc + 1) * FF + n] : 0.f;
            sW[u] = pk2(f0, f1);
        }
    }
    __syncthreads();

    v8f acc0 = {}, acc1 = {}, acc2 = {}, acc3 = {};
    const float* ar = A + (long)m * HH;

#pragma unroll
    for (int k0 = 0; k0 < 64; k0 += 32) {
        const int kb0 = k0 + khalf * 16;
        const int kb1 = kb0 + 8;
        v8u av;
#pragma unroll
        for (int i = 0; i < 4; ++i) av[i]     = pk2(ar[kb0 + 2 * i], ar[kb0 + 2 * i + 1]);
#pragma unroll
        for (int i = 0; i < 4; ++i) av[4 + i] = pk2(ar[kb1 + 2 * i], ar[kb1 + 2 * i + 1]);
        v16bf a = __builtin_bit_cast(v16bf, av);

        const unsigned* wrow = &sW[lrow * 32 + (k0 >> 1) + khalf * 8];
        v16bf b0 = load_frag8(wrow +    0);
        v16bf b1 = load_frag8(wrow +  512);
        v16bf b2 = load_frag8(wrow + 1024);
        v16bf b3 = load_frag8(wrow + 1536);
        acc0 = __builtin_amdgcn_wmma_f32_16x16x32_bf16(false, a, false, b0, (short)0, acc0, false, false);
        acc1 = __builtin_amdgcn_wmma_f32_16x16x32_bf16(false, a, false, b1, (short)0, acc1, false, false);
        acc2 = __builtin_amdgcn_wmma_f32_16x16x32_bf16(false, a, false, b2, (short)0, acc2, false, false);
        acc3 = __builtin_amdgcn_wmma_f32_16x16x32_bf16(false, a, false, b3, (short)0, acc3, false, false);
    }

    const int srow = row0 + 8 * khalf;
    const int n0   = col0 + lrow;
#pragma unroll
    for (int r = 0; r < 8; ++r) {
        int rr = srow + r;
        if (rr < CC) {
            float* orow = xd + (long)rr * FF;
            if (n0      < FF) orow[n0     ] = acc0[r] + bias[n0     ];
            if (n0 + 16 < FF) orow[n0 + 16] = acc1[r] + bias[n0 + 16];
            if (n0 + 32 < FF) orow[n0 + 32] = acc2[r] + bias[n0 + 32];
            if (n0 + 48 < FF) orow[n0 + 48] = acc3[r] + bias[n0 + 48];
        }
    }
}

// ---- graph kernels -------------------------------------------------------

__global__ void degree_kernel(const int* __restrict__ dst, const int* __restrict__ cluster,
                              float* __restrict__ deg, float* __restrict__ deg_c) {
    int e = blockIdx.x * blockDim.x + threadIdx.x;
    if (e < EE) {
        int d = dst[e];
        atomicAdd(&deg[d], 1.0f);
        atomicAdd(&deg_c[cluster[d]], 1.0f);
    }
}

__global__ void count_kernel(const int* __restrict__ cluster, float* __restrict__ cnt) {
    int i = blockIdx.x * blockDim.x + threadIdx.x;
    if (i < NN) atomicAdd(&cnt[cluster[i]], 1.0f);
}

__global__ void finalize_deg_kernel(const float* __restrict__ deg,
                                    float* __restrict__ dinv, float* __restrict__ invd, int n) {
    int i = blockIdx.x * blockDim.x + threadIdx.x;
    if (i < n) {
        float d = deg[i] + 1.0f;
        dinv[i] = rsqrtf(d);
        invd[i] = 1.0f / d;
    }
}

// agg[dst] += h0[src] * dinv[src]*dinv[dst]   (fine graph)
__global__ void enc_prop_kernel(const int* __restrict__ src, const int* __restrict__ dst,
                                const float* __restrict__ dinv,
                                const float* __restrict__ h0, float* __restrict__ agg) {
    long idx = (long)blockIdx.x * blockDim.x + threadIdx.x;
    if (idx >= (long)EE * HH) return;
    int e = (int)(idx >> 6);
    int f = (int)(idx & 63);
    int s = src[e], d = dst[e];
    float norm = dinv[s] * dinv[d];
    atomicAdd(&agg[(long)d * HH + f], h0[(long)s * HH + f] * norm);
}

// h = relu(agg + h0/deg + b_enc); pool: xc_sum[cluster[i]] += h
__global__ void enc_finish_pool_kernel(const float* __restrict__ agg, const float* __restrict__ h0,
                                       const float* __restrict__ invd, const float* __restrict__ b_enc,
                                       const int* __restrict__ cluster, float* __restrict__ xc_sum) {
    long idx = (long)blockIdx.x * blockDim.x + threadIdx.x;
    if (idx >= (long)NN * HH) return;
    int i = (int)(idx >> 6);
    int f = (int)(idx & 63);
    float v = agg[idx] + h0[idx] * invd[i] + b_enc[f];
    v = fmaxf(v, 0.0f);
    atomicAdd(&xc_sum[(long)cluster[i] * HH + f], v);
}

__global__ void xc_final_kernel(float* __restrict__ x_c, const float* __restrict__ cnt) {
    long idx = (long)blockIdx.x * blockDim.x + threadIdx.x;
    if (idx >= (long)CC * HH) return;
    int c = (int)(idx >> 6);
    x_c[idx] = x_c[idx] / fmaxf(cnt[c], 1.0f);
}

// agg_c[dst_c] += x_c[src_c] * dinvc[src_c]*dinvc[dst_c]
__global__ void coarse_prop_kernel(const int* __restrict__ src, const int* __restrict__ dst,
                                   const int* __restrict__ cluster, const float* __restrict__ dinvc,
                                   const float* __restrict__ x_c, float* __restrict__ agg_c) {
    long idx = (long)blockIdx.x * blockDim.x + threadIdx.x;
    if (idx >= (long)EE * HH) return;
    int e = (int)(idx >> 6);
    int f = (int)(idx & 63);
    int sc = cluster[src[e]], dc = cluster[dst[e]];
    float norm = dinvc[sc] * dinvc[dc];
    atomicAdd(&agg_c[(long)dc * HH + f], x_c[(long)sc * HH + f] * norm);
}

__global__ void coarse_finish_kernel(float* __restrict__ agg_c, const float* __restrict__ x_c,
                                     const float* __restrict__ invdc) {
    long idx = (long)blockIdx.x * blockDim.x + threadIdx.x;
    if (idx >= (long)CC * HH) return;
    int c = (int)(idx >> 6);
    agg_c[idx] += x_c[idx] * invdc[c];
}

// unpool: out[i] = x_d[cluster[i]]   (x_d is L2-resident; writes stream at HBM rate)
__global__ void unpool_kernel(const float* __restrict__ xd, const int* __restrict__ cluster,
                              float* __restrict__ out) {
    int f = blockIdx.x * blockDim.x + threadIdx.x;
    int i = blockIdx.y;
    if (f < FF) out[(long)i * FF + f] = xd[(long)cluster[i] * FF + f];
}

// ---- launcher ------------------------------------------------------------

extern "C" void kernel_launch(void* const* d_in, const int* in_sizes, int n_in,
                              void* d_out, int out_size, void* d_ws, size_t ws_size,
                              hipStream_t stream) {
    const float* x       = (const float*)d_in[0];
    const int*   eidx    = (const int*)  d_in[1];
    const int*   cluster = (const int*)  d_in[3];
    const float* W_enc   = (const float*)d_in[4];
    const float* b_enc   = (const float*)d_in[5];
    const float* W_dec   = (const float*)d_in[6];
    const float* b_dec   = (const float*)d_in[7];
    const int* src = eidx;
    const int* dst = eidx + EE;
    float* out = (float*)d_out;

    char* wsp = (char*)d_ws;
    auto alloc = [&](size_t nfloat) {
        float* p = (float*)wsp;
        wsp += ((nfloat * sizeof(float) + 255) / 256) * 256;
        return p;
    };
    float* h0     = alloc((size_t)NN * HH);
    float* agg    = alloc((size_t)NN * HH);
    float* deg    = alloc(NN);
    float* dinv   = alloc(NN);
    float* invd   = alloc(NN);
    float* cnt    = alloc(CC);
    float* deg_c  = alloc(CC);
    float* dinvc  = alloc(CC);
    float* invdc  = alloc(CC);
    float* x_c    = alloc((size_t)CC * HH);
    float* agg_c  = alloc((size_t)CC * HH);
    float* x_d    = alloc((size_t)CC * FF);

    hipMemsetAsync(deg,   0, NN * sizeof(float), stream);
    hipMemsetAsync(deg_c, 0, CC * sizeof(float), stream);
    hipMemsetAsync(cnt,   0, CC * sizeof(float), stream);
    hipMemsetAsync(agg,   0, (size_t)NN * HH * sizeof(float), stream);
    hipMemsetAsync(x_c,   0, (size_t)CC * HH * sizeof(float), stream);
    hipMemsetAsync(agg_c, 0, (size_t)CC * HH * sizeof(float), stream);

    // encoder GEMM (WMMA bf16)
    enc_gemm_kernel<<<(NN + 127) / 128, 256, 0, stream>>>(x, W_enc, h0);

    // degrees / counts
    degree_kernel<<<(EE + 255) / 256, 256, 0, stream>>>(dst, cluster, deg, deg_c);
    count_kernel<<<(NN + 255) / 256, 256, 0, stream>>>(cluster, cnt);
    finalize_deg_kernel<<<(NN + 255) / 256, 256, 0, stream>>>(deg, dinv, invd, NN);
    finalize_deg_kernel<<<(CC + 255) / 256, 256, 0, stream>>>(deg_c, dinvc, invdc, CC);

    // encoder propagate + finish + pool
    long ef = (long)EE * HH;
    enc_prop_kernel<<<(unsigned)((ef + 255) / 256), 256, 0, stream>>>(src, dst, dinv, h0, agg);
    long nf = (long)NN * HH;
    enc_finish_pool_kernel<<<(unsigned)((nf + 255) / 256), 256, 0, stream>>>(
        agg, h0, invd, b_enc, cluster, x_c);
    long cf = (long)CC * HH;
    xc_final_kernel<<<(unsigned)((cf + 255) / 256), 256, 0, stream>>>(x_c, cnt);

    // coarse propagate + finish
    coarse_prop_kernel<<<(unsigned)((ef + 255) / 256), 256, 0, stream>>>(
        src, dst, cluster, dinvc, x_c, agg_c);
    coarse_finish_kernel<<<(unsigned)((cf + 255) / 256), 256, 0, stream>>>(agg_c, x_c, invdc);

    // decoder GEMM (WMMA bf16) + bias
    dim3 dgrid((CC + 127) / 128, (FF + 63) / 64);
    dec_gemm_kernel<<<dgrid, 256, 0, stream>>>(agg_c, W_dec, b_dec, x_d);

    // unpool gather -> output
    dim3 ugrid((FF + 255) / 256, NN);
    unpool_kernel<<<ugrid, 256, 0, stream>>>(x_d, cluster, out);
}
